// Transformer_53042846105923
// MI455X (gfx1250) — compile-verified, matrix operand order checked by reading, not run on hardware
//
#include <hip/hip_runtime.h>
#include <hip/hip_bf16.h>

// Problem constants (match reference)
constexpr int kB   = 8;
constexpr int kS   = 1024;
constexpr int kD   = 256;
constexpr int kH   = 8;
constexpr int kDH  = 32;     // head dim
constexpr int kL   = 4;
constexpr int kV   = 100;
constexpr int kDFF = 1024;
constexpr int kM   = kB * kS;   // 8192 rows

typedef __attribute__((ext_vector_type(16))) _Float16 v16h;
typedef __attribute__((ext_vector_type(8)))  float    v8f;

union Frag16 {
    v16h v;
    unsigned u[8];
    _Float16 h[16];
};

// ---- CDNA5 async copy helpers -------------------------------------------------
__device__ __forceinline__ void async_load_lds_b128(void* lds_ptr, const void* gptr)
{
    unsigned lds_off = (unsigned)(uintptr_t)lds_ptr;          // low 32 bits = LDS offset
    unsigned long long ga = (unsigned long long)(uintptr_t)gptr;
    asm volatile("global_load_async_to_lds_b128 %0, %1, off"
                 :: "v"(lds_off), "v"(ga)
                 : "memory");
}

__device__ __forceinline__ void wait_asynccnt0()
{
#if __has_builtin(__builtin_amdgcn_s_wait_asynccnt)
    __builtin_amdgcn_s_wait_asynccnt(0);
#else
    asm volatile("s_wait_asynccnt 0x0" ::: "memory");
#endif
}

// ---------------------------------------------------------------------------
// Generic WMMA GEMM:  C[M,N] = A[M,K](f16) * B[K,N](f16) + bias  (+res) (relu)
// Block: 256 threads (8 waves). Block tile 128x128. Wave tile 32x64 (8 WMMAs/step).
// A tile staged with GLOBAL_LOAD_ASYNC_TO_LDS_B128 (no transpose needed);
// B tile staged synchronously with an n-major transpose for K-contiguous frags.
// ---------------------------------------------------------------------------
template <int RELU, int RES, int OUTH>
__global__ __launch_bounds__(256) void gemm_f16_wmma(
    const _Float16* __restrict__ A, const _Float16* __restrict__ Bm,
    const float* __restrict__ bias, const float* __restrict__ res,
    float* __restrict__ Cf, _Float16* __restrict__ Ch,
    int M, int K, int N)
{
    constexpr int AS = 40;   // LDS row pitch (halfs) for A tile (128x32)
    constexpr int BS = 40;   // LDS row pitch (halfs) for B tile (128x32, n-major)
    __shared__ __align__(16) _Float16 lA[128 * AS];
    __shared__ __align__(16) _Float16 lB[128 * BS];

    const int tid  = threadIdx.x;
    const int wave = tid >> 5, lane = tid & 31;
    const int lg   = lane >> 4, l15 = lane & 15;
    const int mw   = wave >> 1, nw = wave & 1;       // 4 x 2 wave grid
    const int m0   = blockIdx.x * 128;
    const int n0   = blockIdx.y * 128;

    v8f acc[2][4] = {};

    // B staging map: 32x128 halfs, 16 per thread (two 8-half rows)
    const int kB_ = tid >> 4;           // 0..15  (+16 on second chunk)
    const int nB  = (tid & 15) * 8;     // 0..120

    for (int kk = 0; kk < K; kk += 32) {
        // ---- stage A tile (128x32) via async copy to LDS ----
        #pragma unroll
        for (int c = 0; c < 2; ++c) {
            int id = c * 256 + tid;          // 512 x 16B chunks
            int mA = id >> 2;                // 0..127
            int kA = (id & 3) * 8;           // 0,8,16,24
            async_load_lds_b128(&lA[mA * AS + kA],
                                A + (size_t)(m0 + mA) * K + kk + kA);
        }
        // ---- stage B tile transposed: lB[n][k] ----
        #pragma unroll
        for (int c = 0; c < 2; ++c) {
            int k = kB_ + c * 16;
            if (n0 + nB + 8 <= N) {
                const _Float16* src = Bm + (size_t)(kk + k) * N + n0 + nB;
                float4 t = *reinterpret_cast<const float4*>(src);
                const _Float16* th = reinterpret_cast<const _Float16*>(&t);
                #pragma unroll
                for (int j = 0; j < 8; ++j) lB[(nB + j) * BS + k] = th[j];
            } else {
                #pragma unroll
                for (int j = 0; j < 8; ++j) {
                    int n = n0 + nB + j;
                    _Float16 vv = (_Float16)0.0f;
                    if (n < N) vv = Bm[(size_t)(kk + k) * N + n];
                    lB[(nB + j) * BS + k] = vv;
                }
            }
        }
        // prefetch next K-slice while this one drains
        if (kk + 32 < K) {
            __builtin_prefetch(A + (size_t)(m0 + (tid >> 1)) * K + kk + 32 + (tid & 1) * 16, 0, 0);
            __builtin_prefetch(Bm + (size_t)(kk + 32 + kB_) * N + n0 + nB, 0, 0);
        }
        wait_asynccnt0();
        __syncthreads();

        // ---- fragments + WMMA (2 A-frags x 4 B-frags = 8 WMMA) ----
        Frag16 af[2];
        #pragma unroll
        for (int u = 0; u < 2; ++u) {
            #pragma unroll
            for (int v = 0; v < 8; ++v) {
                int ko = ((v < 4) ? 2 * v : 2 * v + 8) + lg * 8;
                af[u].u[v] = *reinterpret_cast<const unsigned*>(
                    &lA[(mw * 32 + u * 16 + l15) * AS + ko]);
            }
        }
        Frag16 bf[4];
        #pragma unroll
        for (int t = 0; t < 4; ++t) {
            int n = nw * 64 + t * 16 + l15;
            #pragma unroll
            for (int v = 0; v < 8; ++v) {
                int ko = ((v < 4) ? 2 * v : 2 * v + 8) + lg * 8;
                bf[t].u[v] = *reinterpret_cast<const unsigned*>(&lB[n * BS + ko]);
            }
        }
        #pragma unroll
        for (int u = 0; u < 2; ++u) {
            #pragma unroll
            for (int t = 0; t < 4; ++t) {
                acc[u][t] = __builtin_amdgcn_wmma_f32_16x16x32_f16(
                    false, af[u].v, false, bf[t].v, (short)0, acc[u][t], false, false);
            }
        }
        __syncthreads();
    }

    // ---- epilogue ----
    #pragma unroll
    for (int u = 0; u < 2; ++u) {
        #pragma unroll
        for (int t = 0; t < 4; ++t) {
            int n = n0 + nw * 64 + t * 16 + l15;
            if (n < N) {
                #pragma unroll
                for (int r = 0; r < 8; ++r) {
                    int m = m0 + mw * 32 + u * 16 + r + lg * 8;
                    float val = acc[u][t][r] + bias[n];
                    if (RES)  val += res[(size_t)m * N + n];
                    if (RELU) val = fmaxf(val, 0.0f);
                    if (OUTH) Ch[(size_t)m * N + n] = (_Float16)val;
                    else      Cf[(size_t)m * N + n] = val;
                }
            }
        }
    }
}

// ---------------------------------------------------------------------------
// Flash-style causal attention. One wave per (b, h, 16-query tile).
// Q,K: [B,S,D] f16.  Vt: [B,H,32,S] f16 (pre-transposed).  Y: [B,S,D] f16.
// ---------------------------------------------------------------------------
__global__ __launch_bounds__(256) void attn_flash_wmma(
    const _Float16* __restrict__ Q, const _Float16* __restrict__ Kk,
    const _Float16* __restrict__ Vt, const int* __restrict__ slen,
    _Float16* __restrict__ Y)
{
    __shared__ __align__(16) _Float16 plds[8 * 16 * 32];
    const int tid = threadIdx.x, wave = tid >> 5, lane = tid & 31;
    const int lg = lane >> 4, l15 = lane & 15;
    const int id = blockIdx.x * 8 + wave;
    const int qt = id & 63, hh = (id >> 6) & 7, b = id >> 9;
    const int q0 = qt * 16;
    int seqlen = slen[b];
    if (seqlen > kS) seqlen = kS;

    // Q fragment (loaded once)
    Frag16 qf;
    {
        const _Float16* qrow = Q + ((size_t)(b * kS) + q0 + l15) * kD + hh * kDH;
        #pragma unroll
        for (int v = 0; v < 8; ++v) {
            int ko = ((v < 4) ? 2 * v : 2 * v + 8) + lg * 8;
            qf.u[v] = *reinterpret_cast<const unsigned*>(qrow + ko);
        }
    }

    v8f o0 = {}, o1 = {};
    float rmax[8], rsum[8];
    #pragma unroll
    for (int r = 0; r < 8; ++r) { rmax[r] = -1e30f; rsum[r] = 0.0f; }

    int kmax = q0 + 16;
    if (kmax > seqlen) kmax = seqlen;
    const float scale = 0.17677669529663687f;   // 1/sqrt(32)
    _Float16* pw = &plds[wave * 512];

    for (int j0 = 0; j0 < kmax; j0 += 32) {
        // S = Q * K^T  (two 16-key subtiles)
        Frag16 kf0, kf1;
        const _Float16* krow0 = Kk + ((size_t)(b * kS) + j0 + l15) * kD + hh * kDH;
        const _Float16* krow1 = krow0 + (size_t)16 * kD;
        #pragma unroll
        for (int v = 0; v < 8; ++v) {
            int ko = ((v < 4) ? 2 * v : 2 * v + 8) + lg * 8;
            kf0.u[v] = *reinterpret_cast<const unsigned*>(krow0 + ko);
            kf1.u[v] = *reinterpret_cast<const unsigned*>(krow1 + ko);
        }
        v8f s0 = {}, s1 = {};
        s0 = __builtin_amdgcn_wmma_f32_16x16x32_f16(false, qf.v, false, kf0.v, (short)0, s0, false, false);
        s1 = __builtin_amdgcn_wmma_f32_16x16x32_f16(false, qf.v, false, kf1.v, (short)0, s1, false, false);

        // online softmax; write P tile to LDS in A-fragment-friendly layout
        #pragma unroll
        for (int r = 0; r < 8; ++r) {
            int qi = q0 + r + 8 * lg;
            int ja = j0 + l15, jb = j0 + 16 + l15;
            bool okq = qi < seqlen;
            bool va = okq && (ja <= qi) && (ja < seqlen);
            bool vb = okq && (jb <= qi) && (jb < seqlen);
            float sa = va ? s0[r] * scale : -1e9f;
            float sb = vb ? s1[r] * scale : -1e9f;
            float mx = fmaxf(sa, sb);
            #pragma unroll
            for (int o = 8; o >= 1; o >>= 1) mx = fmaxf(mx, __shfl_xor(mx, o, 16));
            float nm = fmaxf(rmax[r], mx);
            float corr = __expf(rmax[r] - nm);
            float pa = va ? __expf(sa - nm) : 0.0f;
            float pb = vb ? __expf(sb - nm) : 0.0f;
            float ps = pa + pb;
            #pragma unroll
            for (int o = 8; o >= 1; o >>= 1) ps += __shfl_xor(ps, o, 16);
            rsum[r] = rsum[r] * corr + ps;
            rmax[r] = nm;
            o0[r] *= corr;
            o1[r] *= corr;
            int m = r + 8 * lg;
            pw[m * 32 + l15]      = (_Float16)pa;
            pw[m * 32 + 16 + l15] = (_Float16)pb;
        }
        __builtin_amdgcn_wave_barrier();

        // O += P * V
        Frag16 pf, vf0, vf1;
        const _Float16* vrow0 = Vt + ((size_t)((b * kH + hh) * kDH) + l15) * kS + j0;
        const _Float16* vrow1 = vrow0 + (size_t)16 * kS;
        #pragma unroll
        for (int v = 0; v < 8; ++v) {
            int ko = ((v < 4) ? 2 * v : 2 * v + 8) + lg * 8;
            pf.u[v]  = *reinterpret_cast<const unsigned*>(&pw[l15 * 32 + ko]);
            vf0.u[v] = *reinterpret_cast<const unsigned*>(vrow0 + ko);
            vf1.u[v] = *reinterpret_cast<const unsigned*>(vrow1 + ko);
        }
        o0 = __builtin_amdgcn_wmma_f32_16x16x32_f16(false, pf.v, false, vf0.v, (short)0, o0, false, false);
        o1 = __builtin_amdgcn_wmma_f32_16x16x32_f16(false, pf.v, false, vf1.v, (short)0, o1, false, false);
        __builtin_amdgcn_wave_barrier();
    }

    #pragma unroll
    for (int r = 0; r < 8; ++r) {
        int qi = q0 + r + 8 * lg;
        float inv = rsum[r] > 0.0f ? 1.0f / rsum[r] : 0.0f;
        _Float16* yrow = Y + ((size_t)(b * kS) + qi) * kD + hh * kDH;
        yrow[l15]      = (_Float16)(o0[r] * inv);
        yrow[16 + l15] = (_Float16)(o1[r] * inv);
    }
}

// ---------------------------------------------------------------------------
// Elementwise / reduction helpers
// ---------------------------------------------------------------------------
__global__ __launch_bounds__(256) void embed_pos_kernel(
    const int* __restrict__ ids, const float* __restrict__ emb,
    float* __restrict__ xf, _Float16* __restrict__ xh)
{
    int idx = blockIdx.x * 256 + threadIdx.x;   // over kM*kD
    int d = idx & (kD - 1);
    int row = idx >> 8;
    int s = row & (kS - 1);
    int tok = ids[row];
    float e = emb[(size_t)tok * kD + d];
    float ex = (d < kD / 2) ? (float)(2 * d) / (float)kD
                            : (float)(2 * (d - kD / 2) + 1) / (float)kD;
    float freq = __expf(-ex * 9.210340371976184f);   // ln(10000)
    float v = e + sinf((float)s * freq);
    xf[idx] = v;
    xh[idx] = (_Float16)v;
}

__global__ __launch_bounds__(256) void layernorm_kernel(
    const float* __restrict__ Hx, const float* __restrict__ g,
    const float* __restrict__ be, float* __restrict__ xf, _Float16* __restrict__ xh)
{
    __shared__ float red[256];
    int row = blockIdx.x, tid = threadIdx.x;
    float v = Hx[(size_t)row * kD + tid];
    red[tid] = v;
    __syncthreads();
    for (int s2 = 128; s2 > 0; s2 >>= 1) {
        if (tid < s2) red[tid] += red[tid + s2];
        __syncthreads();
    }
    float mean = red[0] * (1.0f / kD);
    __syncthreads();
    float dv = v - mean;
    red[tid] = dv * dv;
    __syncthreads();
    for (int s2 = 128; s2 > 0; s2 >>= 1) {
        if (tid < s2) red[tid] += red[tid + s2];
        __syncthreads();
    }
    float var = red[0] * (1.0f / kD);
    float y = dv * rsqrtf(var + 1e-5f) * g[tid] + be[tid];
    xf[(size_t)row * kD + tid] = y;
    xh[(size_t)row * kD + tid] = (_Float16)y;
}

__global__ __launch_bounds__(256) void transpose_v_kernel(
    const _Float16* __restrict__ Vh, _Float16* __restrict__ Vt)
{
    int idx = blockIdx.x * 256 + threadIdx.x;   // over kM*kD
    int c = idx & (kD - 1);
    int row = idx >> 8;                          // b*kS + s
    int s = row & (kS - 1), b = row >> 10;
    int hh = c >> 5, d = c & 31;
    Vt[(((size_t)b * kH + hh) * kDH + d) * kS + s] = Vh[idx];
}

__global__ __launch_bounds__(256) void f32_to_f16_kernel(
    const float* __restrict__ src, _Float16* __restrict__ dst, int n)
{
    int i = blockIdx.x * 256 + threadIdx.x;
    if (i < n) dst[i] = (_Float16)src[i];
}

// ---------------------------------------------------------------------------
// Host-side orchestration
// ---------------------------------------------------------------------------
extern "C" void kernel_launch(void* const* d_in, const int* in_sizes, int n_in,
                              void* d_out, int out_size, void* d_ws, size_t ws_size,
                              hipStream_t stream)
{
    (void)in_sizes; (void)n_in; (void)out_size; (void)ws_size;

    const int*   ids  = (const int*)d_in[0];
    const int*   slen = (const int*)d_in[1];
    const float* emb  = (const float*)d_in[2];
    const float* Wq   = (const float*)d_in[3];  const float* bq  = (const float*)d_in[4];
    const float* Wk   = (const float*)d_in[5];  const float* bk  = (const float*)d_in[6];
    const float* Wv   = (const float*)d_in[7];  const float* bv  = (const float*)d_in[8];
    const float* Wo   = (const float*)d_in[9];  const float* bo  = (const float*)d_in[10];
    const float* W1   = (const float*)d_in[11]; const float* b1  = (const float*)d_in[12];
    const float* W2   = (const float*)d_in[13]; const float* b2  = (const float*)d_in[14];
    const float* g1   = (const float*)d_in[15]; const float* be1 = (const float*)d_in[16];
    const float* g2   = (const float*)d_in[17]; const float* be2 = (const float*)d_in[18];
    const float* Wc   = (const float*)d_in[19]; const float* bc  = (const float*)d_in[20];

    char* p = (char*)d_ws;
    auto carve = [&](size_t bytes) -> void* {
        void* q = (void*)p;
        p += (bytes + 255) & ~(size_t)255;
        return q;
    };

    float*    xf32 = (float*)   carve((size_t)kM * kD * 4);
    float*    hf32 = (float*)   carve((size_t)kM * kD * 4);
    _Float16* xf16 = (_Float16*)carve((size_t)kM * kD * 2);
    _Float16* qf16 = (_Float16*)carve((size_t)kM * kD * 2);
    _Float16* kf16 = (_Float16*)carve((size_t)kM * kD * 2);
    _Float16* vf16 = (_Float16*)carve((size_t)kM * kD * 2);
    _Float16* vt   = (_Float16*)carve((size_t)kM * kD * 2);
    _Float16* yatt = (_Float16*)carve((size_t)kM * kD * 2);
    _Float16* ff1  = (_Float16*)carve((size_t)kM * kDFF * 2);
    _Float16* wqh  = (_Float16*)carve((size_t)kL * kD * kD * 2);
    _Float16* wkh  = (_Float16*)carve((size_t)kL * kD * kD * 2);
    _Float16* wvh  = (_Float16*)carve((size_t)kL * kD * kD * 2);
    _Float16* woh  = (_Float16*)carve((size_t)kL * kD * kD * 2);
    _Float16* w1h  = (_Float16*)carve((size_t)kL * kD * kDFF * 2);
    _Float16* w2h  = (_Float16*)carve((size_t)kL * kDFF * kD * 2);
    _Float16* wch  = (_Float16*)carve((size_t)kD * kV * 2);

    // ---- weight conversions to f16 ----
    {
        int n = kL * kD * kD;
        int g = (n + 255) / 256;
        f32_to_f16_kernel<<<g, 256, 0, stream>>>(Wq, wqh, n);
        f32_to_f16_kernel<<<g, 256, 0, stream>>>(Wk, wkh, n);
        f32_to_f16_kernel<<<g, 256, 0, stream>>>(Wv, wvh, n);
        f32_to_f16_kernel<<<g, 256, 0, stream>>>(Wo, woh, n);
        int n2 = kL * kD * kDFF;
        int g2n = (n2 + 255) / 256;
        f32_to_f16_kernel<<<g2n, 256, 0, stream>>>(W1, w1h, n2);
        f32_to_f16_kernel<<<g2n, 256, 0, stream>>>(W2, w2h, n2);
        int n3 = kD * kV;
        f32_to_f16_kernel<<<(n3 + 255) / 256, 256, 0, stream>>>(Wc, wch, n3);
    }

    // ---- embedding + positional ----
    embed_pos_kernel<<<kM * kD / 256, 256, 0, stream>>>(ids, emb, xf32, xf16);

    const dim3 blk(256);
    const dim3 grD(kM / 128, (kD + 127) / 128);     // N=256 -> 2 col-blocks
    const dim3 grF(kM / 128, (kDFF + 127) / 128);   // N=1024 -> 8
    const dim3 grV(kM / 128, (kV + 127) / 128);     // N=100 -> 1

    for (int l = 0; l < kL; ++l) {
        const _Float16* wq_l = wqh + (size_t)l * kD * kD;
        const _Float16* wk_l = wkh + (size_t)l * kD * kD;
        const _Float16* wv_l = wvh + (size_t)l * kD * kD;
        const _Float16* wo_l = woh + (size_t)l * kD * kD;
        const _Float16* w1_l = w1h + (size_t)l * kD * kDFF;
        const _Float16* w2_l = w2h + (size_t)l * kDFF * kD;

        // Q/K/V projections (f16 out)
        gemm_f16_wmma<0, 0, 1><<<grD, blk, 0, stream>>>(
            xf16, wq_l, bq + l * kD, nullptr, nullptr, qf16, kM, kD, kD);
        gemm_f16_wmma<0, 0, 1><<<grD, blk, 0, stream>>>(
            xf16, wk_l, bk + l * kD, nullptr, nullptr, kf16, kM, kD, kD);
        gemm_f16_wmma<0, 0, 1><<<grD, blk, 0, stream>>>(
            xf16, wv_l, bv + l * kD, nullptr, nullptr, vf16, kM, kD, kD);

        // transpose V for attention B-fragments
        transpose_v_kernel<<<kM * kD / 256, 256, 0, stream>>>(vf16, vt);

        // flash attention -> yatt (f16)
        attn_flash_wmma<<<kB * kH * (kS / 16) / 8, blk, 0, stream>>>(
            qf16, kf16, vt, slen, yatt);

        // output projection + residual (f32 out)
        gemm_f16_wmma<0, 1, 0><<<grD, blk, 0, stream>>>(
            yatt, wo_l, bo + l * kD, xf32, hf32, nullptr, kM, kD, kD);

        // LN1 -> xf32/xf16
        layernorm_kernel<<<kM, blk, 0, stream>>>(hf32, g1 + l * kD, be1 + l * kD, xf32, xf16);

        // FFN: relu(x@W1+b1) -> ff1 (f16)
        gemm_f16_wmma<1, 0, 1><<<grF, blk, 0, stream>>>(
            xf16, w1_l, b1 + l * kDFF, nullptr, nullptr, ff1, kM, kD, kDFF);

        // ff1@W2 + b2 + residual (f32 out)
        gemm_f16_wmma<0, 1, 0><<<grD, blk, 0, stream>>>(
            ff1, w2_l, b2 + l * kD, xf32, hf32, nullptr, kM, kDFF, kD);

        // LN2 -> xf32/xf16
        layernorm_kernel<<<kM, blk, 0, stream>>>(hf32, g2 + l * kD, be2 + l * kD, xf32, xf16);
    }

    // classifier: logits = x@Wc + bc  (f32 to d_out, N=100 guarded)
    gemm_f16_wmma<0, 0, 0><<<grV, blk, 0, stream>>>(
        xf16, wch, bc, nullptr, (float*)d_out, nullptr, kM, kD, kV);
}
